// ReBasedLinearAttention_49615462203434
// MI455X (gfx1250) — matrix operand order repaired and checked by reading, not compiled, not run on hardware
//
#include <hip/hip_runtime.h>
#include <hip/hip_bf16.h>

typedef _Float16 v16h __attribute__((ext_vector_type(16)));
typedef _Float16 v8h  __attribute__((ext_vector_type(8)));
typedef float    v8f  __attribute__((ext_vector_type(8)));

constexpr int BB = 2, LL = 2048, DD = 1024, HH = 16, FD = 16, HD = 64;
constexpr int MM = BB * LL;          // 4096 token rows
constexpr float QK_SCALE = 0.25f;    // FD^-0.5
constexpr float EPS = 1e-5f;

__device__ inline v8f vzero8f() {
  v8f r;
#pragma unroll
  for (int i = 0; i < 8; ++i) r[i] = 0.0f;
  return r;
}

// ---------------------------------------------------------------------------
// Generic WMMA GEMM: out(64x64 tile per block) = A[M,K] (f32 or f16) * W[K,N] (f32)
// 4 waves / block, each wave owns a 16x64 strip; K-loop in steps of 32 with
// LDS staging (A tile row-major, W tile transposed so B-fragments are
// contiguous 32B LDS reads). MODE selects the store layout:
//   0: Q  -> f16 [b,h,l,fd], scaled by QK_SCALE
//   1: K  -> f16 [b,h,l,fd]
//   2: Vt -> f16 [b,h,hd,l]  (transposed for contiguous B-fragment loads)
//   3: f32 row-major [M,N]   (final output projection)
// ---------------------------------------------------------------------------
template <int MODE, typename AT>
__global__ __launch_bounds__(128)
void gemm16_wmma(const AT* __restrict__ A, const float* __restrict__ W,
                 void* __restrict__ outp, int M, int K, int N) {
  __shared__ _Float16 Xs[64][40]  __attribute__((aligned(16)));
  __shared__ _Float16 WsT[64][40] __attribute__((aligned(16)));
  const int t = threadIdx.x;
  const int lane = t & 31;
  const int wave = t >> 5;
  const int row0 = blockIdx.y * 64;
  const int n0 = blockIdx.x * 64;

  v8f acc[4];
#pragma unroll
  for (int n = 0; n < 4; ++n) acc[n] = vzero8f();

  const int arow = t >> 1;        // 0..63  (A tile row)
  const int acg = (t & 1) * 16;   // 0/16   (A tile col group)
  const int wk = t >> 2;          // 0..31  (W tile k row)
  const int wcg = (t & 3) * 16;   // 0..48  (W tile col group)

  for (int k0 = 0; k0 < K; k0 += 32) {
    // prefetch next K tile
    if (k0 + 32 < K) {
      __builtin_prefetch(A + (size_t)(row0 + arow) * K + (k0 + 32 + acg), 0, 1);
      __builtin_prefetch(W + (size_t)(k0 + 32 + wk) * N + (n0 + wcg), 0, 1);
    }
    {  // stage A tile (convert to f16)
      const AT* src = A + (size_t)(row0 + arow) * K + (k0 + acg);
#pragma unroll
      for (int i = 0; i < 16; ++i) Xs[arow][acg + i] = (_Float16)src[i];
    }
    {  // stage W tile, transposed: WsT[n][k]
      const float* src = W + (size_t)(k0 + wk) * N + (n0 + wcg);
#pragma unroll
      for (int i = 0; i < 16; ++i) WsT[wcg + i][wk] = (_Float16)src[i];
    }
    __syncthreads();

    // A fragment: 16x32 f16, K = (lane>=16?16:0)+h
    const int rowL = wave * 16 + (lane & 15);
    const int baseA = (lane < 16) ? 0 : 16;
    v8h alo = *(const v8h*)&Xs[rowL][baseA];
    v8h ahi = *(const v8h*)&Xs[rowL][baseA + 8];
    v16h afrag;
#pragma unroll
    for (int i = 0; i < 8; ++i) { afrag[i] = alo[i]; afrag[8 + i] = ahi[i]; }

    // B fragments: K = (lane>=16?16:0)+h, N = lane&15  -> contiguous in WsT[n][k]
    const int kbB = (lane < 16) ? 0 : 16;
#pragma unroll
    for (int n = 0; n < 4; ++n) {
      const int colL = n * 16 + (lane & 15);
      v8h blo = *(const v8h*)&WsT[colL][kbB];
      v8h bhi = *(const v8h*)&WsT[colL][kbB + 8];
      v16h bfrag;
#pragma unroll
      for (int i = 0; i < 8; ++i) { bfrag[i] = blo[i]; bfrag[8 + i] = bhi[i]; }
      acc[n] = __builtin_amdgcn_wmma_f32_16x16x32_f16(false, afrag, false, bfrag,
                                                      (short)0, acc[n], false, false);
    }
    __syncthreads();
  }

  // store: C/D layout M = r + (lane>=16?8:0), N = lane&15
#pragma unroll
  for (int n = 0; n < 4; ++n) {
#pragma unroll
    for (int r = 0; r < 8; ++r) {
      const int Mi = (lane < 16) ? r : (r + 8);
      const int gr = row0 + wave * 16 + Mi;
      const int gc = n0 + n * 16 + (lane & 15);
      float v = acc[n][r];
      if constexpr (MODE == 0 || MODE == 1) {
        if constexpr (MODE == 0) v *= QK_SCALE;
        const int b = gr / LL, l = gr % LL;
        const int head = gc >> 4, fd = gc & 15;
        ((_Float16*)outp)[(((size_t)(b * HH + head) * LL) + l) * FD + fd] = (_Float16)v;
      } else if constexpr (MODE == 2) {
        const int b = gr / LL, l = gr % LL;
        const int head = gc / HD, hd = gc % HD;
        ((_Float16*)outp)[(((size_t)(b * HH + head) * HD) + hd) * LL + l] = (_Float16)v;
      } else {
        ((float*)outp)[(size_t)gr * N + gc] = v;
      }
    }
  }
}

// ---------------------------------------------------------------------------
// Causal rebased (quadratic-kernel) attention, flash-style.
// One wave per 32-query block (two 16-row M-tiles) per (b,h), so every K/V
// fragment load is amortized over 2x the WMMAs. Per 32-key chunk:
//   4 WMMA  -> S_u = Q_u Kt (u=0,1; K-dim 16 zero-padded to 32), square+mask
//   LDS round-trip re-lays S (C-layout) into two A-fragments (16x32 f16)
//   2 WMMA  -> z_u += S_u @ ones (acc_z[u][r] holds z[row M], N-replicated)
//   8 WMMA  -> O_u += S_u @ V    (HD = 64, V fragments loaded once, shared)
// With q0 = 32*qb the chunk loop runs c = 0..qb and every sub-tile satisfies
// key16 <= q0+31, so the causal mask alone handles the diagonal: branch-free.
// Final: O[M][hd] / (z[M]+eps), lane-local because z replicates across N.
// ---------------------------------------------------------------------------
__global__ __launch_bounds__(32)
void rebased_attn_wmma(const _Float16* __restrict__ Qh, const _Float16* __restrict__ Kh,
                       const _Float16* __restrict__ Vt, _Float16* __restrict__ Ob) {
  __shared__ _Float16 Ssh[32][40] __attribute__((aligned(16)));
  const int lane = threadIdx.x & 31;
  const int qb = blockIdx.x, head = blockIdx.y, b = blockIdx.z;
  const int bh = b * HH + head;
  const int q0 = qb * 32;

  // Q fragments: fd 0..15 in K-slots 0..15, K-slots 16..31 zero
  const int baseA = (lane < 16) ? 0 : 8;
  v16h aq[2];
#pragma unroll
  for (int u = 0; u < 2; ++u) {
    const int qrow = q0 + u * 16 + (lane & 15);
    const _Float16* qp = Qh + ((size_t)bh * LL + qrow) * FD;
    v8h qlo = *(const v8h*)qp;
    v8h qhi = *(const v8h*)(qp + 8);
#pragma unroll
    for (int i = 0; i < 8; ++i) { aq[u][i] = (lane < 16) ? qlo[i] : (_Float16)0; aq[u][8 + i] = (lane < 16) ? qhi[i] : (_Float16)0; }
  }

  v16h ones;
#pragma unroll
  for (int i = 0; i < 16; ++i) ones[i] = (_Float16)1.0f;

  v8f acc_o[2][4];
  v8f acc_z[2];
#pragma unroll
  for (int u = 0; u < 2; ++u) {
    acc_z[u] = vzero8f();
#pragma unroll
    for (int n = 0; n < 4; ++n) acc_o[u][n] = vzero8f();
  }

  for (int c = 0; c <= qb; ++c) {
    __syncthreads();
    // build 32x32 squared+masked score tile in LDS (four 16x16 WMMAs)
#pragma unroll
    for (int tt = 0; tt < 2; ++tt) {
      const int key16 = c * 32 + tt * 16;
      const int colS = tt * 16 + (lane & 15);
      v16h kb;
      if (lane < 16) {
        // B-fragment of K^T: fd rows 0..15 live in lanes 0..15, N = key col
        const _Float16* kp = Kh + ((size_t)bh * LL + (key16 + lane)) * FD;
        kb = *(const v16h*)kp;
      } else {
#pragma unroll
        for (int i = 0; i < 16; ++i) kb[i] = (_Float16)0;  // padded fd 16..31
      }
      const int kg = key16 + (lane & 15);
#pragma unroll
      for (int u = 0; u < 2; ++u) {
        v8f s = __builtin_amdgcn_wmma_f32_16x16x32_f16(false, aq[u], false, kb,
                                                       (short)0, vzero8f(), false, false);
#pragma unroll
        for (int r = 0; r < 8; ++r) {
          const int Mi = (lane < 16) ? r : (r + 8);
          float sv = s[r];
          sv = sv * sv;                          // rebased quadratic feature map
          if (kg > q0 + u * 16 + Mi) sv = 0.0f;  // causal mask
          Ssh[u * 16 + Mi][colS] = (_Float16)sv;
        }
      }
    }
    __syncthreads();

    // S as two A-fragments (16 queries x 32 keys each) + z accumulation
    v16h as[2];
#pragma unroll
    for (int u = 0; u < 2; ++u) {
      const int rowS = u * 16 + (lane & 15);
      const int baseS = (lane < 16) ? 0 : 16;
      v8h slo = *(const v8h*)&Ssh[rowS][baseS];
      v8h shi = *(const v8h*)&Ssh[rowS][baseS + 8];
#pragma unroll
      for (int i = 0; i < 8; ++i) { as[u][i] = slo[i]; as[u][8 + i] = shi[i]; }
      acc_z[u] = __builtin_amdgcn_wmma_f32_16x16x32_f16(false, as[u], false, ones,
                                                        (short)0, acc_z[u], false, false);
    }
    // O += S @ V, V fragments loaded once and shared by both M-tiles
#pragma unroll
    for (int n = 0; n < 4; ++n) {
      const int hd = n * 16 + (lane & 15);
      const int kb32 = c * 32 + ((lane < 16) ? 0 : 16);
      const _Float16* vp = Vt + ((size_t)bh * HD + hd) * LL + kb32;
      v16h vb = *(const v16h*)vp;
#pragma unroll
      for (int u = 0; u < 2; ++u) {
        acc_o[u][n] = __builtin_amdgcn_wmma_f32_16x16x32_f16(false, as[u], false, vb,
                                                             (short)0, acc_o[u][n], false, false);
      }
    }
  }

  // normalize + store merged-head f16 output [b, l, h*HD + hd]
#pragma unroll
  for (int u = 0; u < 2; ++u) {
#pragma unroll
    for (int n = 0; n < 4; ++n) {
#pragma unroll
      for (int r = 0; r < 8; ++r) {
        const int Mi = (lane < 16) ? r : (r + 8);
        const int q = q0 + u * 16 + Mi;
        const float o = acc_o[u][n][r] / (acc_z[u][r] + EPS);
        Ob[((size_t)(b * LL + q)) * (HH * HD) + head * HD + n * 16 + (lane & 15)] = (_Float16)o;
      }
    }
  }
}

extern "C" void kernel_launch(void* const* d_in, const int* in_sizes, int n_in,
                              void* d_out, int out_size, void* d_ws, size_t ws_size,
                              hipStream_t stream) {
  (void)in_sizes; (void)n_in; (void)out_size; (void)ws_size;
  const float* X  = (const float*)d_in[0];
  const float* Wq = (const float*)d_in[1];
  const float* Wk = (const float*)d_in[2];
  const float* Wv = (const float*)d_in[3];
  const float* Wo = (const float*)d_in[4];
  float* out = (float*)d_out;

  char* ws = (char*)d_ws;
  _Float16* Qh = (_Float16*)(ws + 0);                     // 2 MB  [b,h,l,fd]
  _Float16* Kh = (_Float16*)(ws + (size_t)(2u << 20));    // 2 MB  [b,h,l,fd]
  _Float16* Vt = (_Float16*)(ws + (size_t)(4u << 20));    // 8 MB  [b,h,hd,l]
  _Float16* Ob = (_Float16*)(ws + (size_t)(12u << 20));   // 8 MB  [b,l,h*hd]

  dim3 blk(128);
  gemm16_wmma<0, float><<<dim3((HH * FD) / 64, MM / 64), blk, 0, stream>>>(X, Wq, Qh, MM, DD, HH * FD);
  gemm16_wmma<1, float><<<dim3((HH * FD) / 64, MM / 64), blk, 0, stream>>>(X, Wk, Kh, MM, DD, HH * FD);
  gemm16_wmma<2, float><<<dim3((HH * HD) / 64, MM / 64), blk, 0, stream>>>(X, Wv, Vt, MM, DD, HH * HD);
  rebased_attn_wmma<<<dim3(LL / 32, HH, BB), dim3(32), 0, stream>>>(Qh, Kh, Vt, Ob);
  gemm16_wmma<3, _Float16><<<dim3(DD / 64, MM / 64), blk, 0, stream>>>(Ob, Wo, out, MM, HH * HD, DD);
}
